// GAT_layer_node_only_86088324481812
// MI455X (gfx1250) — matrix operand order repaired and checked by reading, not compiled, not run on hardware
//
#include <hip/hip_runtime.h>
#include <hip/hip_bf16.h>

#define N_NODES   100000
#define N_EDGES   1600000
#define IN_DIM    128
#define EDGE_DIM  32
#define H_HEADS   4
#define C_DIM     32
#define HC        128   // H_HEADS * C_DIM

typedef __attribute__((ext_vector_type(16))) __bf16 v16bf;
typedef __attribute__((ext_vector_type(8)))  float  v8f;

// ---------------------------------------------------------------------------
// K0: init accumulators. d_out doubles as the segment-sum accumulator.
// ---------------------------------------------------------------------------
__global__ void k_init(float* __restrict__ out, float* __restrict__ amax,
                       float* __restrict__ asum) {
  int i = blockIdx.x * blockDim.x + threadIdx.x;
  if (i < N_NODES * HC) out[i] = 0.0f;
  if (i < N_NODES * H_HEADS) { amax[i] = -3.0e38f; asum[i] = 0.0f; }
}

// ---------------------------------------------------------------------------
// K0b: pack W (row-major f32 [128,128]) into bf16 B-fragments in WMMA lane
// order. Fragment (nt,s) covers K rows [s*32, s*32+32), N cols [nt*16,nt*16+16).
// Lane L holds col = nt*16 + (L&15), K = s*32 + 8*(L>>4) + (j&7) + 16*(j>>3), j=0..15 packed
// contiguously -> one 32-byte span per lane, coalesced across lanes.
// ---------------------------------------------------------------------------
__global__ void k_pack_w(const float* __restrict__ W, __bf16* __restrict__ Wp) {
  int i = blockIdx.x * blockDim.x + threadIdx.x;   // 32 frags * 32 lanes * 16 j
  if (i >= 32 * 32 * 16) return;
  int j    = i & 15;
  int lane = (i >> 4) & 31;
  int frag = i >> 9;            // nt*4 + s
  int s  = frag & 3;
  int nt = frag >> 2;
  int k  = s * 32 + 8 * (lane >> 4) + (j & 7) + 16 * (j >> 3);
  int n  = nt * 16 + (lane & 15);
  Wp[i] = (__bf16)W[(size_t)k * HC + n];
}

// ---------------------------------------------------------------------------
// K1: x = point_attr @ W via v_wmma_f32_16x16x32_bf16, one wave per 16 rows.
// A 16x32 bf16 layout: lane(0-15) row=lane, K in {0..7,16..23}; lanes 16-31
// same rows, K in {8..15,24..31}.  B fragments come pre-packed from k_pack_w
// (2x global_load_b128 per fragment).  Result stored as bf16 (feeds the
// bandwidth-bound message pass at half the gather traffic).
// ---------------------------------------------------------------------------
__global__ void k_node_gemm(const float* __restrict__ A,
                            const __bf16* __restrict__ Wp,
                            __bf16* __restrict__ xb) {
  const int lane  = threadIdx.x & 31;
  const int wave  = blockIdx.x * (blockDim.x >> 5) + (threadIdx.x >> 5);
  const int tiles = N_NODES / 16;
  if (wave >= tiles) return;               // uniform per wave: EXEC stays all-1s
  const int m0  = wave * 16;
  const int row = m0 + (lane & 15);
  const int hi  = lane >> 4;               // 0 or 1
  const int col = lane & 15;

  v16bf afrag[4];
  #pragma unroll
  for (int s = 0; s < 4; ++s) {
    const float* ap0 = A + (size_t)row * IN_DIM + s * 32 + hi * 8;
    const float* ap1 = ap0 + 16;
    #pragma unroll
    for (int j = 0; j < 8; ++j) {
      afrag[s][j]     = (__bf16)ap0[j];
      afrag[s][j + 8] = (__bf16)ap1[j];
    }
  }

  #pragma unroll
  for (int nt = 0; nt < 8; ++nt) {
    v8f acc = {0.f, 0.f, 0.f, 0.f, 0.f, 0.f, 0.f, 0.f};
    #pragma unroll
    for (int s = 0; s < 4; ++s) {
      // packed fragment: 16 contiguous bf16 per lane, 32B aligned
      v16bf bfrag = *(const v16bf*)(Wp + (size_t)(((nt << 2) | s) * 32 + lane) * 16);
      acc = __builtin_amdgcn_wmma_f32_16x16x32_bf16(
                false, afrag[s], false, bfrag, (short)0, acc, false, false);
    }
    #pragma unroll
    for (int r = 0; r < 8; ++r)
      xb[(size_t)(m0 + r + 8 * hi) * HC + nt * 16 + col] = (__bf16)acc[r];
  }
}

// ---------------------------------------------------------------------------
// K2: a_src[n,h], a_dst[n,h] = <x[n,h,:], att_src/att_dst[h,:]>
// ---------------------------------------------------------------------------
__global__ void k_node_att(const __bf16* __restrict__ xb,
                           const float* __restrict__ att_src,
                           const float* __restrict__ att_dst,
                           float* __restrict__ a_src, float* __restrict__ a_dst) {
  int i = blockIdx.x * blockDim.x + threadIdx.x;
  if (i >= N_NODES * H_HEADS) return;
  int n = i >> 2, h = i & 3;
  const __bf16* xp = xb + (size_t)n * HC + h * C_DIM;
  float s1 = 0.f, s2 = 0.f;
  #pragma unroll
  for (int c = 0; c < C_DIM; ++c) {
    float xv = (float)xp[c];
    s1 += xv * att_src[h * C_DIM + c];
    s2 += xv * att_dst[h * C_DIM + c];
  }
  a_src[i] = s1;
  a_dst[i] = s2;
}

// ---------------------------------------------------------------------------
// K3: fold W_edge with att_edge: w_eh[h,k] = sum_c W_edge[k, h*C+c]*att_edge[h,c]
// (a_edge = edge_attr @ w_eh^T -> avoids materializing [E,128] e-tensor)
// ---------------------------------------------------------------------------
__global__ void k_fold_edge(const float* __restrict__ W_edge,
                            const float* __restrict__ att_edge,
                            float* __restrict__ w_eh) {
  int i = threadIdx.x;                     // one block of 128
  if (i >= H_HEADS * EDGE_DIM) return;
  int h = i >> 5, k = i & 31;
  float s = 0.f;
  #pragma unroll
  for (int c = 0; c < C_DIM; ++c)
    s += W_edge[(size_t)k * HC + h * C_DIM + c] * att_edge[h * C_DIM + c];
  w_eh[h * EDGE_DIM + k] = s;
}

// ---------------------------------------------------------------------------
// K4: per edge: alpha = lrelu0.2(a_src[src]+a_dst[dst]+edge_attr@w_eh),
// atomic float max into amax[dst]  (global_atomic_max_num_f32).
// ---------------------------------------------------------------------------
__global__ void k_edge_alpha(const long long* __restrict__ ei,
                             const float* __restrict__ edge_attr,
                             const float* __restrict__ w_eh,
                             const float* __restrict__ a_src,
                             const float* __restrict__ a_dst,
                             float* __restrict__ alpha, float* __restrict__ amax) {
  int e = blockIdx.x * blockDim.x + threadIdx.x;
  if (e >= N_EDGES) return;
  int src = (int)ei[e];
  int dst = (int)ei[N_EDGES + e];
  float ea[EDGE_DIM];
  const float4* p = (const float4*)(edge_attr + (size_t)e * EDGE_DIM);
  #pragma unroll
  for (int q = 0; q < 8; ++q) {
    float4 t = p[q];
    ea[4 * q] = t.x; ea[4 * q + 1] = t.y; ea[4 * q + 2] = t.z; ea[4 * q + 3] = t.w;
  }
  #pragma unroll
  for (int h = 0; h < H_HEADS; ++h) {
    float s = 0.f;
    #pragma unroll
    for (int k = 0; k < EDGE_DIM; ++k) s += ea[k] * w_eh[h * EDGE_DIM + k];
    float al = a_src[src * H_HEADS + h] + a_dst[dst * H_HEADS + h] + s;
    al = al > 0.f ? al : 0.2f * al;
    alpha[(size_t)e * H_HEADS + h] = al;
    __hip_atomic_fetch_max(&amax[dst * H_HEADS + h], al,
                           __ATOMIC_RELAXED, __HIP_MEMORY_SCOPE_AGENT);
  }
}

// ---------------------------------------------------------------------------
// K5: expa = exp(alpha - amax[dst]); atomic add into asum[dst].
// ---------------------------------------------------------------------------
__global__ void k_edge_exp(const long long* __restrict__ ei,
                           float* __restrict__ alpha,
                           const float* __restrict__ amax,
                           float* __restrict__ asum) {
  int e = blockIdx.x * blockDim.x + threadIdx.x;
  if (e >= N_EDGES) return;
  int dst = (int)ei[N_EDGES + e];
  #pragma unroll
  for (int h = 0; h < H_HEADS; ++h) {
    float al = alpha[(size_t)e * H_HEADS + h];
    float ex = __expf(al - amax[dst * H_HEADS + h]);
    alpha[(size_t)e * H_HEADS + h] = ex;
    __hip_atomic_fetch_add(&asum[dst * H_HEADS + h], ex,
                           __ATOMIC_RELAXED, __HIP_MEMORY_SCOPE_AGENT);
  }
}

// ---------------------------------------------------------------------------
// K6: message pass (bandwidth-bound hot loop). One 128-thread block per edge
// so src/dst/expa addresses are wave-uniform -> scalar (s_load) path; per
// thread: bf16 gather of x[src], scale, native f32 atomic add into out[dst].
// ---------------------------------------------------------------------------
__global__ void k_message(const long long* __restrict__ ei,
                          const float* __restrict__ expa,
                          const float* __restrict__ asum,
                          const __bf16* __restrict__ xb,
                          float* __restrict__ out) {
  const int e = blockIdx.x;             // uniform per block
  const int c = threadIdx.x;            // 0..127
  const int h = c >> 5;
  const int src = (int)ei[e];
  const int dst = (int)ei[N_EDGES + e];
  float w = expa[(size_t)e * H_HEADS + h] / (asum[dst * H_HEADS + h] + 1e-16f);
  float v = (float)xb[(size_t)src * HC + c] * w;
  __hip_atomic_fetch_add(&out[(size_t)dst * HC + c], v,
                         __ATOMIC_RELAXED, __HIP_MEMORY_SCOPE_AGENT);
}

// ---------------------------------------------------------------------------
// K7: bias + LayerNorm(128) + leaky_relu(0.01), one wave per node, in place.
// ---------------------------------------------------------------------------
__global__ void k_finalize(float* __restrict__ out, const float* __restrict__ bias,
                           const float* __restrict__ gamma,
                           const float* __restrict__ beta) {
  int n = blockIdx.x * (blockDim.x >> 5) + (threadIdx.x >> 5);
  if (n >= N_NODES) return;
  int lane = threadIdx.x & 31;
  float v[4];
  float sum = 0.f;
  #pragma unroll
  for (int j = 0; j < 4; ++j) {
    int c = lane + 32 * j;
    v[j] = out[(size_t)n * HC + c] + bias[c];
    sum += v[j];
  }
  #pragma unroll
  for (int o = 16; o > 0; o >>= 1) sum += __shfl_xor(sum, o, 32);
  float mean = sum * (1.0f / 128.0f);
  float var = 0.f;
  #pragma unroll
  for (int j = 0; j < 4; ++j) { float d = v[j] - mean; var += d * d; }
  #pragma unroll
  for (int o = 16; o > 0; o >>= 1) var += __shfl_xor(var, o, 32);
  float rstd = rsqrtf(var * (1.0f / 128.0f) + 1e-5f);
  #pragma unroll
  for (int j = 0; j < 4; ++j) {
    int c = lane + 32 * j;
    float o_ = (v[j] - mean) * rstd * gamma[c] + beta[c];
    o_ = o_ > 0.f ? o_ : 0.01f * o_;
    out[(size_t)n * HC + c] = o_;
  }
}

// ---------------------------------------------------------------------------
extern "C" void kernel_launch(void* const* d_in, const int* in_sizes, int n_in,
                              void* d_out, int out_size, void* d_ws, size_t ws_size,
                              hipStream_t stream) {
  const float*     point_attr = (const float*)d_in[0];
  const long long* edge_index = (const long long*)d_in[1];   // int64 per reference
  const float*     edge_attr  = (const float*)d_in[2];
  const float*     W          = (const float*)d_in[3];
  const float*     att_src    = (const float*)d_in[4];
  const float*     att_dst    = (const float*)d_in[5];
  const float*     W_edge     = (const float*)d_in[6];
  const float*     att_edge   = (const float*)d_in[7];
  const float*     bias       = (const float*)d_in[8];
  const float*     ln_gamma   = (const float*)d_in[9];
  const float*     ln_beta    = (const float*)d_in[10];
  float* out = (float*)d_out;

  // Workspace carve-up (~58 MB total)
  char* p = (char*)d_ws;
  auto carve = [&](size_t bytes) -> char* {
    char* r = p;
    p += (bytes + 255) & ~(size_t)255;
    return r;
  };
  __bf16* xb    = (__bf16*)carve((size_t)N_NODES * HC * sizeof(__bf16));
  float*  alpha = (float*) carve((size_t)N_EDGES * H_HEADS * sizeof(float));
  float*  a_src = (float*) carve((size_t)N_NODES * H_HEADS * sizeof(float));
  float*  a_dst = (float*) carve((size_t)N_NODES * H_HEADS * sizeof(float));
  float*  amax  = (float*) carve((size_t)N_NODES * H_HEADS * sizeof(float));
  float*  asum  = (float*) carve((size_t)N_NODES * H_HEADS * sizeof(float));
  float*  w_eh  = (float*) carve((size_t)H_HEADS * EDGE_DIM * sizeof(float));
  __bf16* Wp    = (__bf16*)carve((size_t)32 * 32 * 16 * sizeof(__bf16)); // packed W

  // K0: init (N*HC = 12.8M elems)
  k_init<<<(N_NODES * HC + 255) / 256, 256, 0, stream>>>(out, amax, asum);

  // K0b: pack W into WMMA B-fragment order (16384 elems)
  k_pack_w<<<(32 * 32 * 16) / 256, 256, 0, stream>>>(W, Wp);

  // K1: node GEMM, 4 waves/block, one wave per 16-row tile (6250 tiles)
  {
    int tiles = N_NODES / 16;
    int wavesPerBlock = 4;
    int grid = (tiles + wavesPerBlock - 1) / wavesPerBlock;
    k_node_gemm<<<grid, wavesPerBlock * 32, 0, stream>>>(point_attr, Wp, xb);
  }

  // K2: per-(node,head) attention coefficients
  k_node_att<<<(N_NODES * H_HEADS + 255) / 256, 256, 0, stream>>>(
      xb, att_src, att_dst, a_src, a_dst);

  // K3: fold edge weight with edge attention vector
  k_fold_edge<<<1, 128, 0, stream>>>(W_edge, att_edge, w_eh);

  // K4: alpha + segment max
  k_edge_alpha<<<N_EDGES / 256, 256, 0, stream>>>(
      edge_index, edge_attr, w_eh, a_src, a_dst, alpha, amax);

  // K5: exp + segment sum
  k_edge_exp<<<N_EDGES / 256, 256, 0, stream>>>(edge_index, alpha, amax, asum);

  // K6: gather-scale-scatter message pass, one block per edge (uniform scalars)
  k_message<<<N_EDGES, 128, 0, stream>>>(edge_index, alpha, asum, xb, out);

  // K7: bias + LayerNorm + LeakyReLU, 8 nodes (waves) per block
  k_finalize<<<N_NODES / 8, 256, 0, stream>>>(out, bias, ln_gamma, ln_beta);
}
